// VQ_EMA_zalando_1099511627878
// MI455X (gfx1250) — matrix operand order repaired and checked by reading, not compiled, not run on hardware
//
#include <hip/hip_runtime.h>
#include <hip/hip_bf16.h>
#include <stdint.h>

typedef __attribute__((ext_vector_type(16))) _Float16     v16h;
typedef __attribute__((ext_vector_type(8)))  float        v8f;
typedef __attribute__((ext_vector_type(4)))  unsigned int v4u;
typedef __attribute__((ext_vector_type(8)))  int          v8i;
typedef __attribute__((ext_vector_type(4)))  int          v4i;

#define N_ROWS 16384   // B*T
#define DIM    512
#define K_CODES 8192
#define DECAY_F 0.99f
#define ONE_MINUS_DECAY 0.01f
#define EPS_F 1e-5f

#define TILE_CODES 16
#define TILE_BYTES (TILE_CODES * DIM * 2)   // 16 KB per f16 B tile

#if __has_builtin(__builtin_amdgcn_sched_barrier)
#define SCHED_FENCE() __builtin_amdgcn_sched_barrier(0)
#else
#define SCHED_FENCE() asm volatile("" ::: "memory")
#endif

// ---------------------------------------------------------------------------
// Kernel 1: convert codebook fp32 -> f16, compute per-code squared L2 norm.
// ---------------------------------------------------------------------------
__global__ void prep_embed_kernel(const float* __restrict__ embed_w,
                                  _Float16* __restrict__ eh,
                                  float* __restrict__ e2) {
    __shared__ float red[256];
    const int k = blockIdx.x;
    const int t = threadIdx.x;
    const float* src = embed_w + (size_t)k * DIM;
    _Float16* dst = eh + (size_t)k * DIM;
    float s = 0.f;
#pragma unroll
    for (int j = 0; j < 2; ++j) {
        int d = t + j * 256;
        float v = src[d];
        dst[d] = (_Float16)v;
        s += v * v;
    }
    red[t] = s;
    __syncthreads();
    for (int off = 128; off > 0; off >>= 1) {
        if (t < off) red[t] += red[t + off];
        __syncthreads();
    }
    if (t == 0) e2[k] = red[0];
}

// ---------------------------------------------------------------------------
// TDM issue: 2D tensor descriptor per CDNA5 ISA D# layout (08_async_tensor).
// Loads one 16-code x 512-feature f16 tile (16 KB) from global into LDS.
// ---------------------------------------------------------------------------
__device__ __forceinline__ void tdm_load_tile(uint64_t gaddr, uint32_t lds_off) {
    v4u g0;
    g0.x = 1u;                                               // count=1
    g0.y = lds_off;                                          // lds_addr
    g0.z = (uint32_t)(gaddr & 0xFFFFFFFFull);                // global_addr lo
    g0.w = (uint32_t)((gaddr >> 32) & 0x01FFFFFFull)         // global_addr hi
         | (2u << 30);                                       // type = 2 (image)
    v8i g1;
    g1[0] = 0x10000;                  // data_size = 1 (2 bytes)
    g1[1] = (int)(512u << 16);        // tensor_dim0[15:0] = 512 (bits 79:48)
    g1[2] = (int)(8192u << 16);       // tensor_dim1[15:0] = 8192 (bits 111:80)
    g1[3] = (int)(512u << 16);        // tile_dim0 = 512 (bits 127:112)
    g1[4] = TILE_CODES;               // tile_dim1 = 16 (bits 143:128)
    g1[5] = 512;                      // tensor_dim0_stride lo32 = 512
    g1[6] = (int)(512u << 16);        // tensor_dim1_stride[15:0] = 512 (255:208)
    g1[7] = 0;                        // tensor_dim1_stride hi
    v4i z4 = {0, 0, 0, 0};
#if defined(__clang_major__) && __clang_major__ >= 23
    v8i z8 = {0, 0, 0, 0, 0, 0, 0, 0};
    __builtin_amdgcn_tensor_load_to_lds(g0, g1, z4, z4, z8, 0);
#else
    __builtin_amdgcn_tensor_load_to_lds(g0, g1, z4, z4, 0);
#endif
}

// ---------------------------------------------------------------------------
// Kernel 2: WMMA distance GEMM + fused argmin, TDM-staged B tiles in LDS.
// 8 waves/block; each wave owns a 16-row x-tile (A in registers). Wave 0
// double-buffers 16-code B tiles into LDS via tensor_load_to_lds; all waves
// run v_wmma_f32_16x16x32_f16 from LDS with a sched_barrier-enforced 2-ahead
// B prefetch and two independent accumulator chains.
// dist_proxy = ||e||^2 - 2<x,e>  (||x||^2 constant per row, dropped)
// ---------------------------------------------------------------------------
__global__ void __launch_bounds__(256, 1)
vq_argmin_kernel(const float* __restrict__ x,
                 const _Float16* __restrict__ eh,
                 const float* __restrict__ e2,
                 int* __restrict__ idx_out) {
    __shared__ _Float16 btile[2][TILE_CODES * DIM];          // 2 x 16 KB

    const int tid  = threadIdx.x;
    const int lane = tid & 31;
    const int wave = tid >> 5;
    const int m    = lane & 15;
    const bool hi  = lane >= 16;
    const int row_base = blockIdx.x * 128 + wave * 16;

    const uint32_t lds_base = (uint32_t)(uintptr_t)(&btile[0][0]);
    const uint64_t ehaddr   = (uint64_t)(uintptr_t)eh;

    // ---- A tile: 16 rows x 512 features, f32 -> f16 fragments in VGPRs ----
    v16h a[16];
    const float* xrow = x + (size_t)(row_base + m) * DIM;
#pragma unroll
    for (int c = 0; c < 16; ++c) {
        const float* bp = xrow + c * 32 + (hi ? 16 : 0);
        float4 f0 = *(const float4*)(bp);
        float4 f1 = *(const float4*)(bp + 4);
        float4 f2 = *(const float4*)(bp + 8);
        float4 f3 = *(const float4*)(bp + 12);
        v16h av;
        av[0]  = (_Float16)f0.x; av[1]  = (_Float16)f0.y;
        av[2]  = (_Float16)f0.z; av[3]  = (_Float16)f0.w;
        av[4]  = (_Float16)f1.x; av[5]  = (_Float16)f1.y;
        av[6]  = (_Float16)f1.z; av[7]  = (_Float16)f1.w;
        av[8]  = (_Float16)f2.x; av[9]  = (_Float16)f2.y;
        av[10] = (_Float16)f2.z; av[11] = (_Float16)f2.w;
        av[12] = (_Float16)f3.x; av[13] = (_Float16)f3.y;
        av[14] = (_Float16)f3.z; av[15] = (_Float16)f3.w;
        a[c] = av;
    }

    float minv[8];
    int   mini[8];
#pragma unroll
    for (int r = 0; r < 8; ++r) { minv[r] = 3.4e38f; mini[r] = 0; }

    const int NT = K_CODES / TILE_CODES;   // 512 tiles

    // Prologue: wave 0 launches the first two tiles via the Tensor Data Mover.
    if (tid < 32) {
        tdm_load_tile(ehaddr, lds_base);
        tdm_load_tile(ehaddr + TILE_BYTES, lds_base + TILE_BYTES);
    }

    for (int kt = 0; kt < NT; ++kt) {
        // TENSORcnt is per-wave; waves 1..7 have cnt==0 so this is free for
        // them. In-order TDM: cnt<=1 means tile kt has fully landed in LDS.
        __builtin_amdgcn_s_wait_tensorcnt(1);
        __syncthreads();

        // B layout (32x16 f16): lane = column n (lane&15), K 0..15 / 16..31.
        const _Float16* bt = &btile[kt & 1][0] + (size_t)m * DIM + (hi ? 16 : 0);

        // Software-pipelined B prefetch: loads for chunk pair c+2 are pinned
        // (sched_barrier) above the WMMAs for chunk pair c, so the wait in
        // front of each WMMA is dscnt<=4 with two WMMAs of latency cover.
        v16h b0 = *(const v16h*)(bt);
        v16h b1 = *(const v16h*)(bt + 32);
        v8f acc0 = {0.f, 0.f, 0.f, 0.f, 0.f, 0.f, 0.f, 0.f};
        v8f acc1 = {0.f, 0.f, 0.f, 0.f, 0.f, 0.f, 0.f, 0.f};
#pragma unroll
        for (int c = 0; c < 16; c += 2) {
            v16h n0 = b0, n1 = b1;
            if (c + 2 < 16) {
                n0 = *(const v16h*)(bt + (c + 2) * 32);
                n1 = *(const v16h*)(bt + (c + 3) * 32);
            }
            SCHED_FENCE();   // keep next-chunk ds_loads above these WMMAs
            acc0 = __builtin_amdgcn_wmma_f32_16x16x32_f16(
                false, a[c], false, b0, (short)0, acc0, false, false);
            acc1 = __builtin_amdgcn_wmma_f32_16x16x32_f16(
                false, a[c + 1], false, b1, (short)0, acc1, false, false);
            SCHED_FENCE();   // keep these WMMAs above the next chunk's loads
            b0 = n0;
            b1 = n1;
        }
        v8f acc = acc0 + acc1;

        const int code = kt * 16 + m;
        const float en = e2[code];
#pragma unroll
        for (int r = 0; r < 8; ++r) {
            float dproxy = en - 2.0f * acc[r];
            bool better = dproxy < minv[r];
            minv[r] = better ? dproxy : minv[r];
            mini[r] = better ? code : mini[r];
        }

        __syncthreads();   // everyone done reading buf[kt&1]
        if (tid < 32 && kt + 2 < NT) {
            // refill the buffer we just consumed with tile kt+2
            tdm_load_tile(ehaddr + (uint64_t)(kt + 2) * TILE_BYTES,
                          lds_base + (uint32_t)(kt & 1) * TILE_BYTES);
        }
    }

    // Cross-lane argmin reduce within each 16-lane half (rows 0..7 / 8..15).
#pragma unroll
    for (int s = 1; s < 16; s <<= 1) {
#pragma unroll
        for (int r = 0; r < 8; ++r) {
            float ov = __shfl_xor(minv[r], s, 32);
            int   oi = __shfl_xor(mini[r], s, 32);
            bool take = (ov < minv[r]) || (ov == minv[r] && oi < mini[r]);
            minv[r] = take ? ov : minv[r];
            mini[r] = take ? oi : mini[r];
        }
    }
    if ((lane & 15) == 0) {
        int rb = row_base + (hi ? 8 : 0);
#pragma unroll
        for (int r = 0; r < 8; ++r) idx_out[rb + r] = mini[r];
    }
}

// ---------------------------------------------------------------------------
// Kernel 3: init accumulators. new_ema_w (in d_out) = DECAY*ema_w,
// counts = 0, n_sum = 0, loss = 0.
// ---------------------------------------------------------------------------
__global__ void init_kernel(const float* __restrict__ ema_w,
                            float* __restrict__ new_emaw,
                            float* __restrict__ counts,
                            float* __restrict__ n_sum,
                            float* __restrict__ loss) {
    int gid = blockIdx.x * blockDim.x + threadIdx.x;
    if (gid < K_CODES * DIM) new_emaw[gid] = DECAY_F * ema_w[gid];
    if (gid < K_CODES) counts[gid] = 0.f;
    if (gid == 0) { *n_sum = 0.f; *loss = 0.f; }
}

// ---------------------------------------------------------------------------
// Kernel 4: scatter counts + dw into new_ema_w via global f32 atomics.
// ---------------------------------------------------------------------------
__global__ void scatter_kernel(const float* __restrict__ x,
                               const int* __restrict__ idx,
                               float* __restrict__ counts,
                               float* __restrict__ new_emaw) {
    const int row = blockIdx.x;
    const int t = threadIdx.x;
    const int code = idx[row];
    if (t == 0) atomicAdd(&counts[code], 1.0f);
    const float* src = x + (size_t)row * DIM;
    float* dst = new_emaw + (size_t)code * DIM;
#pragma unroll
    for (int j = 0; j < 2; ++j) {
        int d = t + j * 256;
        atomicAdd(&dst[d], ONE_MINUS_DECAY * src[d]);
    }
}

// ---------------------------------------------------------------------------
// Kernel 5: straight-through output + commitment loss.
// ---------------------------------------------------------------------------
__global__ void quant_loss_kernel(const float* __restrict__ x,
                                  const float* __restrict__ embed_w,
                                  const int* __restrict__ idx,
                                  float* __restrict__ quant,
                                  float* __restrict__ loss) {
    __shared__ float red[256];
    const int row = blockIdx.x;
    const int t = threadIdx.x;
    const int code = idx[row];
    const float* xr = x + (size_t)row * DIM;
    const float* er = embed_w + (size_t)code * DIM;
    float* qr = quant + (size_t)row * DIM;
    float s = 0.f;
#pragma unroll
    for (int j = 0; j < 2; ++j) {
        int d = t + j * 256;
        float xv = xr[d];
        float qv = er[d];
        qr[d] = xv + (qv - xv);
        float df = xv - qv;
        s += df * df;
    }
    red[t] = s;
    __syncthreads();
    for (int off = 128; off > 0; off >>= 1) {
        if (t < off) red[t] += red[t + off];
        __syncthreads();
    }
    if (t == 0)
        atomicAdd(loss, red[0] * (0.25f / (float)(N_ROWS * DIM)));
}

// ---------------------------------------------------------------------------
// Kernel 6: pre-smoothing cluster size + total sum n (pre_cs overwrites counts)
// ---------------------------------------------------------------------------
__global__ void cs_sum_kernel(const float* __restrict__ ema_cs,
                              float* __restrict__ counts,
                              float* __restrict__ n_sum) {
    __shared__ float red[256];
    const int k = blockIdx.x * 256 + threadIdx.x;
    const int t = threadIdx.x;
    float pre = ema_cs[k] * DECAY_F + ONE_MINUS_DECAY * counts[k];
    counts[k] = pre;
    red[t] = pre;
    __syncthreads();
    for (int off = 128; off > 0; off >>= 1) {
        if (t < off) red[t] += red[t + off];
        __syncthreads();
    }
    if (t == 0) atomicAdd(n_sum, red[0]);
}

// ---------------------------------------------------------------------------
// Kernel 7: finalize. new_cs = (pre+eps)/(n+K*eps)*n ; new_embed = ema_w/new_cs
// ---------------------------------------------------------------------------
__global__ void finalize_kernel(const float* __restrict__ pre_cs,
                                const float* __restrict__ n_sum,
                                const float* __restrict__ new_emaw,
                                float* __restrict__ new_embed,
                                float* __restrict__ new_cs) {
    int gid = blockIdx.x * blockDim.x + threadIdx.x;
    int k = gid >> 9;  // / DIM
    float n = *n_sum;
    float ncs = (pre_cs[k] + EPS_F) / (n + (float)K_CODES * EPS_F) * n;
    new_embed[gid] = new_emaw[gid] / ncs;
    if ((gid & (DIM - 1)) == 0) new_cs[k] = ncs;
}

// ---------------------------------------------------------------------------
extern "C" void kernel_launch(void* const* d_in, const int* in_sizes, int n_in,
                              void* d_out, int out_size, void* d_ws, size_t ws_size,
                              hipStream_t stream) {
    const float* x       = (const float*)d_in[0];   // [32,512,512]
    const float* embed_w = (const float*)d_in[1];   // [8192,512]
    const float* ema_w   = (const float*)d_in[2];   // [8192,512]
    const float* ema_cs  = (const float*)d_in[3];   // [8192]

    float* out = (float*)d_out;
    float* quant     = out;                                        // 8388608
    float* loss      = out + (size_t)N_ROWS * DIM;                 // 1
    float* new_embed = loss + 1;                                   // 4194304
    float* new_emaw  = new_embed + (size_t)K_CODES * DIM;          // 4194304
    float* new_cs    = new_emaw + (size_t)K_CODES * DIM;           // 8192

    char* ws = (char*)d_ws;
    _Float16* eh = (_Float16*)ws;                                  // 8 MB
    float* e2    = (float*)(ws + (size_t)K_CODES * DIM * 2);       // 32 KB
    int*   idx   = (int*)((char*)e2 + (size_t)K_CODES * 4);        // 64 KB
    float* counts= (float*)((char*)idx + (size_t)N_ROWS * 4);      // 32 KB
    float* n_sum = (float*)((char*)counts + (size_t)K_CODES * 4);  // 4 B

    prep_embed_kernel<<<K_CODES, 256, 0, stream>>>(embed_w, eh, e2);
    vq_argmin_kernel<<<N_ROWS / 128, 256, 0, stream>>>(x, eh, e2, idx);
    init_kernel<<<(K_CODES * DIM) / 256, 256, 0, stream>>>(ema_w, new_emaw,
                                                           counts, n_sum, loss);
    scatter_kernel<<<N_ROWS, 256, 0, stream>>>(x, idx, counts, new_emaw);
    quant_loss_kernel<<<N_ROWS, 256, 0, stream>>>(x, embed_w, idx, quant, loss);
    cs_sum_kernel<<<K_CODES / 256, 256, 0, stream>>>(ema_cs, counts, n_sum);
    finalize_kernel<<<(K_CODES * DIM) / 256, 256, 0, stream>>>(counts, n_sum,
                                                               new_emaw, new_embed,
                                                               new_cs);
}
